// ConvLSTMModel_59399397703680
// MI455X (gfx1250) — matrix operand order repaired and checked by reading, not compiled, not run on hardware
//
#include <hip/hip_runtime.h>
#include <cstdint>
#include <cstddef>

typedef __attribute__((ext_vector_type(16))) __bf16 v16bf;
typedef __attribute__((ext_vector_type(8)))  float  v8f;

#define W48   48
#define H48   48
#define PLANE (48*48)
#define CPAD  96
#define KTOT  864      // 9 taps * 96 padded channels
#define KSTEPS 27      // 864 / 32
#define LDSX  50       // 48 + 2 halo columns

struct U32x8 { uint4 a, b; };   // 32 bytes == v16bf

__device__ __forceinline__ unsigned short f2bf(float f) {
  unsigned u = __builtin_bit_cast(unsigned, f);
  unsigned r = u + 0x7FFFu + ((u >> 16) & 1u);   // round-to-nearest-even
  return (unsigned short)(r >> 16);
}
__device__ __forceinline__ float sigm(float x) { return 1.f / (1.f + __expf(-x)); }
__device__ __forceinline__ float tanh_fast(float x) {
  float t = __expf(-2.f * fabsf(x));
  float r = (1.f - t) / (1.f + t);
  return copysignf(r, x);
}

// ---------------- weight prep: fp32 [M, CINT, 3, 3] -> bf16 [M, 864] (k = (ky*3+kx)*96 + c)
__global__ void prep_weights(const float* __restrict__ w, unsigned short* __restrict__ A,
                             int M, int CINT) {
  int idx = blockIdx.x * 256 + threadIdx.x;
  if (idx >= M * KTOT) return;
  int m = idx / KTOT, k = idx % KTOT;
  int tap = k / CPAD, c = k % CPAD;
  int ky = tap / 3, kx = tap % 3;
  float v = 0.f;
  if (c < CINT) v = w[((m * CINT + c) * 3 + ky) * 3 + kx];
  A[idx] = f2bf(v);
}

__global__ void zero_f32(float* p, int n) {
  int i = blockIdx.x * 256 + threadIdx.x;
  if (i < n) p[i] = 0.f;
}

// ---------------- fused ConvLSTM step (conv -> gates -> state update), one block per (b, y) row
template <int HID, int CIN>
__global__ __launch_bounds__(256)
void convlstm_step(const float* __restrict__ xin, int xb_stride,
                   const float* __restrict__ hprev,
                   float* __restrict__ hout,
                   float* __restrict__ cbuf,
                   const unsigned short* __restrict__ A,   // bf16 [COUT4, 864]
                   const float* __restrict__ bias,
                   float* __restrict__ seq_out) {          // b==0 h dump (layer1), or null
  constexpr int COUT4 = 4 * HID;
  constexpr int MT    = COUT4 / 128;     // M-tiles per wave: 2 (layer0) / 1 (layer1)
  constexpr int CINT  = CIN + HID;

  extern __shared__ char smem[];
  unsigned short* s_in = (unsigned short*)smem;                 // [3][LDSX][CPAD] bf16
  float*          s_g  = (float*)(smem + 3 * LDSX * CPAD * 2);  // [COUT4][48] gates

  const int b = blockIdx.x / H48;
  const int y = blockIdx.x % H48;
  const int tid = threadIdx.x;

  // ---- Phase 1: stage conv_in rows y-1..y+1 (x halo, channel pad) as bf16 into LDS
  for (int item = tid; item < 3 * LDSX * (CPAD / 2); item += 256) {
    int cpair = item % (CPAD / 2);
    int x     = (item / (CPAD / 2)) % LDSX;
    int row   =  item / (LDSX * (CPAD / 2));
    int iy = y + row - 1;
    int ix = x - 1;
    unsigned packed = 0;
    if (iy >= 0 && iy < H48 && ix >= 0 && ix < W48) {
#pragma unroll
      for (int e = 0; e < 2; ++e) {
        int c = cpair * 2 + e;
        float v = 0.f;
        if (c < CIN)       v = xin[b * xb_stride + c * PLANE + iy * W48 + ix];
        else if (c < CINT) v = hprev[(b * HID + (c - CIN)) * PLANE + iy * W48 + ix];
        packed |= ((unsigned)f2bf(v)) << (16 * e);
      }
    }
    ((unsigned*)s_in)[item] = packed;
  }
  __syncthreads();

  // ---- Phase 2: implicit-GEMM conv, D[COUT4 x 48] += A[COUT4 x 864] * im2col[864 x 48]
  const int wave = tid >> 5;
  const int lane = tid & 31;
  const int lm   = lane & 15;
  const int lg   = lane >> 4;

  const v8f vzero = {0.f, 0.f, 0.f, 0.f, 0.f, 0.f, 0.f, 0.f};
  v8f acc[MT][3];
#pragma unroll
  for (int i = 0; i < MT; ++i)
#pragma unroll
    for (int j = 0; j < 3; ++j) acc[i][j] = vzero;

  for (int ks = 0; ks < KSTEPS; ++ks) {
    const int k0 = ks * 32;

    // A fragments (16x32 bf16): lanes 0-15 -> K k0+[0..15]; lanes 16-31 -> K k0+[16..31]
    v16bf afr[MT];
#pragma unroll
    for (int i = 0; i < MT; ++i) {
      int m  = (wave * MT + i) * 16 + lm;
      int kb = k0 + lg * 16;
      const uint4* p = (const uint4*)(A + m * KTOT + kb);
      U32x8 u; u.a = p[0]; u.b = p[1];
      afr[i] = __builtin_bit_cast(v16bf, u);
    }

    // B fragments (32x16 bf16): lane group g holds K k0+g*16 .. +15 (16 contiguous channels)
    v16bf bfr[3];
    {
      int kk  = k0 + lg * 16;
      int tap = kk / CPAD;
      int c0  = kk % CPAD;
      int ky  = tap / 3, kx = tap % 3;
#pragma unroll
      for (int j = 0; j < 3; ++j) {
        int x = j * 16 + lm;                  // image x of column n; LDS col = x + kx
        const uint4* p = (const uint4*)(s_in + (ky * LDSX + x + kx) * CPAD + c0);
        U32x8 u; u.a = p[0]; u.b = p[1];
        bfr[j] = __builtin_bit_cast(v16bf, u);
      }
    }

#pragma unroll
    for (int i = 0; i < MT; ++i)
#pragma unroll
      for (int j = 0; j < 3; ++j)
        acc[i][j] = __builtin_amdgcn_wmma_f32_16x16x32_bf16(
            false, afr[i], false, bfr[j], (short)0, acc[i][j], false, false);
  }

  // ---- Phase 3: gates (+bias) -> LDS for cross-wave i/f/o/g recombination
#pragma unroll
  for (int i = 0; i < MT; ++i) {
    int mbase = (wave * MT + i) * 16 + lg * 8;
#pragma unroll
    for (int r = 0; r < 8; ++r) {
      float bv = bias[mbase + r];
#pragma unroll
      for (int j = 0; j < 3; ++j)
        s_g[(mbase + r) * W48 + j * 16 + lm] = acc[i][j][r] + bv;
    }
  }
  __syncthreads();

  // ---- Phase 4: LSTM state update (c in place, h double-buffered)
  for (int idx = tid; idx < HID * W48; idx += 256) {
    int ch = idx / W48;
    int xx = idx % W48;
    float gi = s_g[ch * W48 + xx];
    float gf = s_g[(HID     + ch) * W48 + xx];
    float go = s_g[(2 * HID + ch) * W48 + xx];
    float gg = s_g[(3 * HID + ch) * W48 + xx];
    int off = (b * HID + ch) * PLANE + y * W48 + xx;
    float cold = cbuf[off];
    float cnew = sigm(gf) * cold + sigm(gi) * tanh_fast(gg);
    float hnew = sigm(go) * tanh_fast(cnew);
    cbuf[off] = cnew;
    hout[off] = hnew;
    if (seq_out != nullptr && b == 0)
      seq_out[ch * PLANE + y * W48 + xx] = hnew;
  }
}

// ---------------- head: raw-memory regrouped dot: out[k] = sum_j seq_flat[k*32+j]*wl[j] + bl
__global__ void head_kernel(const float* __restrict__ seq, const float* __restrict__ wl,
                            const float* __restrict__ bl, float* __restrict__ out, int n) {
  __shared__ float sw[32];
  if (threadIdx.x < 32) sw[threadIdx.x] = wl[threadIdx.x];
  __syncthreads();
  int k = blockIdx.x * 256 + threadIdx.x;
  if (k >= n) return;
  const float4* p = (const float4*)(seq + (size_t)k * 32);
  float s = 0.f;
#pragma unroll
  for (int q = 0; q < 8; ++q) {
    float4 v = p[q];
    s += v.x * sw[q * 4 + 0] + v.y * sw[q * 4 + 1] + v.z * sw[q * 4 + 2] + v.w * sw[q * 4 + 3];
  }
  out[k] = s + bl[0];
}

extern "C" void kernel_launch(void* const* d_in, const int* in_sizes, int n_in,
                              void* d_out, int out_size, void* d_ws, size_t ws_size,
                              hipStream_t stream) {
  const float* x  = (const float*)d_in[0];
  const float* w0 = (const float*)d_in[1];
  const float* b0 = (const float*)d_in[2];
  const float* w1 = (const float*)d_in[3];
  const float* b1 = (const float*)d_in[4];
  const float* wl = (const float*)d_in[5];
  const float* bl = (const float*)d_in[6];
  float* out = (float*)d_out;

  char* ws = (char*)d_ws;
  size_t off = 0;
  auto alloc = [&](size_t bytes) -> void* {
    void* p = ws + off;
    off += (bytes + 255) & ~(size_t)255;
    return p;
  };
  // states first: h0a,c0,h1a,c1 contiguous (all sizes multiple of 256B) -> single zero pass
  float* h0a = (float*)alloc((size_t)8 * 64 * PLANE * 4);
  float* c0  = (float*)alloc((size_t)8 * 64 * PLANE * 4);
  float* h1a = (float*)alloc((size_t)8 * 32 * PLANE * 4);
  float* c1  = (float*)alloc((size_t)8 * 32 * PLANE * 4);
  float* h0b = (float*)alloc((size_t)8 * 64 * PLANE * 4);
  float* h1b = (float*)alloc((size_t)8 * 32 * PLANE * 4);
  unsigned short* A0 = (unsigned short*)alloc((size_t)256 * KTOT * 2);
  unsigned short* A1 = (unsigned short*)alloc((size_t)128 * KTOT * 2);
  float* seq = (float*)alloc((size_t)32 * 32 * PLANE * 4);   // [T,32,H,W] of batch 0

  int nz = 8 * 64 * PLANE * 2 + 8 * 32 * PLANE * 2;          // h0a + c0 + h1a + c1 (floats)
  zero_f32<<<(nz + 255) / 256, 256, 0, stream>>>(h0a, nz);

  prep_weights<<<(256 * KTOT + 255) / 256, 256, 0, stream>>>(w0, A0, 256, 65);
  prep_weights<<<(128 * KTOT + 255) / 256, 256, 0, stream>>>(w1, A1, 128, 96);

  const size_t lds0 = 3 * LDSX * CPAD * 2 + (size_t)256 * W48 * 4;  // 28800 + 49152
  const size_t lds1 = 3 * LDSX * CPAD * 2 + (size_t)128 * W48 * 4;  // 28800 + 24576

  float* h0c = h0a; float* h0n = h0b;
  float* h1c = h1a; float* h1n = h1b;
  for (int t = 0; t < 32; ++t) {
    // layer0: input x_t ([B,T,1,H,W]), per-b stride T*PLANE
    convlstm_step<64, 1><<<8 * H48, 256, lds0, stream>>>(
        x + (size_t)t * PLANE, 32 * PLANE, h0c, h0n, c0, A0, b0, nullptr);
    // layer1: input = fresh h0_t, per-b stride 64*PLANE; dump b==0 h into seq[t]
    convlstm_step<32, 64><<<8 * H48, 256, lds1, stream>>>(
        h0n, 64 * PLANE, h1c, h1n, c1, A1, b1, seq + (size_t)t * 32 * PLANE);
    float* tmp = h0c; h0c = h0n; h0n = tmp;
    tmp = h1c; h1c = h1n; h1n = tmp;
  }

  const int NOUT = 32 * PLANE;  // 73728
  head_kernel<<<(NOUT + 255) / 256, 256, 0, stream>>>(seq, wl, bl, out, NOUT);
}
